// GaugeConvolutionBlock_87101936763182
// MI455X (gfx1250) — compile-verified, matrix operand order checked by reading, not run on hardware
//
#include <hip/hip_runtime.h>
#include <hip/hip_bf16.h>
#include <math.h>

// GaugeConvolutionBlock, restructured:
//   agg[b,i,e] = sum_{j,f} GELU(A1[b,i,f]+A2[b,j,f]+gb1[f]) * W2T[b,j,e,f] + bias[b,e]
// with W2T[b,j,e,f] = sum_d gw2[f, d*D+e] * q[b,j,d].
// Avoids materializing U (B,S,S,D,D): ~69 GFLOP -> ~2.6 GFLOP.
// GEMMs: v_wmma_f32_16x16x32_bf16 (wave32, fp32 accum).
// k_gauge stages its block-shared A1/A2 rows into LDS with
// GLOBAL_LOAD_ASYNC_TO_LDS_B128 (ASYNCcnt path), B-operand streams from L2.

#define BB 2
#define SS 128
#define DD 64
#define FF 256
#define BS (BB * SS)   // 256 rows total
#define NC (DD * FF)   // 16384: per-(b,j) W2T block

typedef __attribute__((ext_vector_type(16))) __bf16 v16bf;
typedef __attribute__((ext_vector_type(8)))  float  v8f;

__device__ __forceinline__ v8f wmma_bf16(v16bf a, v16bf b, v8f c) {
  // (neg_a, A, neg_b, B, c_mod, C, reuse_a, reuse_b)
  return __builtin_amdgcn_wmma_f32_16x16x32_bf16(false, a, false, b, (short)0, c,
                                                 false, false);
}

// ISA 7.12.2: 16-bit A-operand (16x32). Lane L: m = L&15, half = L>>4.
// Packed slot t (0..15) holds K = (t>=8 ? 16 : 0) + half*8 + (t&7).
// B-operand symmetric (lane = column n, same K striping).
__device__ __forceinline__ int kmap(int t, int half) {
  return ((t >> 3) << 4) + half * 8 + (t & 7);
}

__device__ __forceinline__ float gelu_erf(float x) {
  return 0.5f * x * (1.0f + erff(x * 0.7071067811865475f));
}

// 16-byte async copy global -> LDS, per active lane (tracked by ASYNCcnt).
__device__ __forceinline__ void async_b128(unsigned lds_off, const void* gaddr) {
  asm volatile("global_load_async_to_lds_b128 %0, %1, off"
               :: "v"(lds_off), "v"((unsigned long long)(size_t)gaddr)
               : "memory");
}
__device__ __forceinline__ void wait_async0() {
  asm volatile("s_wait_asynccnt 0x0" ::: "memory");
}

// ---------------------------------------------------------------- pack kernels
// GtT[(e*FF + f)*64 + d] = bf16( gw2[f*4096 + d*64 + e] )
__global__ void k_pack_gt(const float* __restrict__ gw2, __bf16* __restrict__ gt) {
  int g = blockIdx.x * 256 + threadIdx.x;   // grid 4096*256
  int c = g >> 6, d = g & 63;
  int e = c >> 8, f = c & 255;
  gt[g] = (__bf16)gw2[f * 4096 + d * 64 + e];
}

__global__ void k_pack_phi(const float* __restrict__ q, __bf16* __restrict__ phib) {
  int g = blockIdx.x * 256 + threadIdx.x;   // grid 64*256 = 16384
  phib[g] = (__bf16)q[g];
}

// ---------------------------------------------------------------- A1/A2 = q @ gw1
__global__ void __launch_bounds__(256) k_a12(const float* __restrict__ q,
                                             const float* __restrict__ gw1,
                                             float* __restrict__ A1,
                                             float* __restrict__ A2) {
  __shared__ float row[DD];
  int bj = blockIdx.x;         // 0..255
  int f  = threadIdx.x;        // 0..255
  if (f < DD) row[f] = q[bj * DD + f];
  __syncthreads();
  float s1 = 0.f, s2 = 0.f;
  for (int d = 0; d < DD; ++d) {
    float r = row[d];
    s1 += r * gw1[d * FF + f];
    s2 += r * gw1[(DD + d) * FF + f];
  }
  A1[bj * FF + f] = s1;
  A2[bj * FF + f] = s2;
}

// ---------------------------------------------------------------- W2T GEMM (WMMA)
// OUT (256 x 16384) = phib (256 x 64) @ GtT^T ; store bf16 row-major.
__global__ void __launch_bounds__(256) k_w2(const __bf16* __restrict__ phib,
                                            const __bf16* __restrict__ gt,
                                            __bf16* __restrict__ w2t) {
  int wave = threadIdx.x >> 5, lane = threadIdx.x & 31;
  int tile = blockIdx.x * 8 + wave;        // 0..16383
  int mt = tile & 15, nt = tile >> 4;
  int m = lane & 15, half = lane >> 4;
  v8f acc;
#pragma unroll
  for (int r = 0; r < 8; ++r) acc[r] = 0.f;
#pragma unroll
  for (int kk = 0; kk < 2; ++kk) {
    v16bf a, b;
#pragma unroll
    for (int t = 0; t < 16; ++t) {
      int k = kk * 32 + kmap(t, half);
      a[t] = phib[(mt * 16 + m) * DD + k];          // A: rows = bj
      b[t] = gt[(size_t)(nt * 16 + m) * DD + k];    // B^T: lane's column, d contiguous
    }
    acc = wmma_bf16(a, b, acc);
  }
#pragma unroll
  for (int r = 0; r < 8; ++r) {
    int row = mt * 16 + r + 8 * half;                // C: M = r + 8*half, N = lane&15
    w2t[(size_t)row * NC + nt * 16 + m] = (__bf16)acc[r];
  }
}

// ---------------------------------------------------------------- agg init (gb2 bias)
__global__ void k_init(const float* __restrict__ q, const float* __restrict__ gb2,
                       float* __restrict__ agg) {
  __shared__ float sbar[DD];
  int b = blockIdx.x, t = threadIdx.x;    // 64 threads
  float s = 0.f;
  for (int j = 0; j < SS; ++j) s += q[(b * SS + j) * DD + t];
  sbar[t] = s;
  __syncthreads();
  float bias = 0.f;
  for (int d = 0; d < DD; ++d) bias += sbar[d] * gb2[d * DD + t];
  for (int i = 0; i < SS; ++i) agg[(b * SS + i) * DD + t] = bias;
}

// ---------------------------------------------------------------- main fused kernel
// Block = (b, i-tile, j-group). A1 rows (16x256) and A2 rows (16x256) staged
// into LDS via async-to-LDS b128 (32 KB contiguous), gb1 folded into sA1.
// 8 waves, each owns 2 j's, 4 e-tile accumulators; GELU A-tiles in registers.
__global__ void __launch_bounds__(256) k_gauge(const float* __restrict__ A1,
                                               const float* __restrict__ A2,
                                               const float* __restrict__ gb1,
                                               const __bf16* __restrict__ w2t,
                                               float* __restrict__ agg) {
  __shared__ float sA1[16 * FF];    // 16 KB: A1 rows of this i-tile (+gb1)
  __shared__ float sA2[16 * FF];    // 16 KB: A2 rows of this j-group
  __shared__ float red[16 * 64];    //  4 KB: cross-wave tile reduction
  int wave = threadIdx.x >> 5, lane = threadIdx.x & 31;
  int jg = blockIdx.x & 7, it = (blockIdx.x >> 3) & 7, b = blockIdx.x >> 6;
  int m = lane & 15, half = lane >> 4;

  // ---- async stage: two flat 16 KB copies (rows are contiguous)
  {
    const char* gA1 = (const char*)(A1 + (b * SS + it * 16) * FF);
    const char* gA2 = (const char*)(A2 + (b * SS + jg * 16) * FF);
    unsigned l1 = (unsigned)(size_t)sA1;
    unsigned l2 = (unsigned)(size_t)sA2;
#pragma unroll
    for (int k = 0; k < 4; ++k) {
      unsigned off = (threadIdx.x + k * 256) * 16;   // bytes
      async_b128(l1 + off, gA1 + off);
      async_b128(l2 + off, gA2 + off);
    }
  }
  for (int x = threadIdx.x; x < 16 * 64; x += 256) red[x] = 0.f;
  wait_async0();
  __syncthreads();
  for (int x = threadIdx.x; x < 16 * FF; x += 256) sA1[x] += gb1[x & (FF - 1)];
  __syncthreads();

  v8f acc[4];
#pragma unroll
  for (int n = 0; n < 4; ++n)
#pragma unroll
    for (int r = 0; r < 8; ++r) acc[n][r] = 0.f;

  const float* a1row = sA1 + m * FF;      // this lane's i-row (gb1 folded)

  for (int jj = 0; jj < 2; ++jj) {
    int jloc = wave * 2 + jj;                       // j within group
    int j = jg * 16 + jloc;
    const float* a2row = sA2 + jloc * FF;
    const __bf16* w2base = w2t + (size_t)(b * SS + j) * NC;
    for (int fc = 0; fc < 8; ++fc) {
      int f0 = fc * 32;
      __builtin_prefetch((const void*)(w2base + f0 + 512), 0, 0);  // global_prefetch_b8
      v16bf a;
#pragma unroll
      for (int t = 0; t < 16; ++t) {
        int f = f0 + kmap(t, half);
        float x = a1row[f] + a2row[f];              // ds_load, 32B-contiguous chunks
        a[t] = (__bf16)gelu_erf(x);
      }
#pragma unroll
      for (int n = 0; n < 4; ++n) {
        const __bf16* col = w2base + (size_t)(n * 16 + m) * FF + f0;
        v16bf bmat;
#pragma unroll
        for (int t = 0; t < 16; ++t) bmat[t] = col[kmap(t, half)];
        acc[n] = wmma_bf16(a, bmat, acc[n]);
      }
    }
  }

  // reduce 8 waves' partial tiles in LDS (ds_add_f32), then add to global agg
#pragma unroll
  for (int n = 0; n < 4; ++n)
#pragma unroll
    for (int r = 0; r < 8; ++r)
      atomicAdd(&red[(r + 8 * half) * 64 + n * 16 + m], acc[n][r]);
  __syncthreads();
  for (int x = threadIdx.x; x < 16 * 64; x += 256) {
    int mi = x >> 6, e = x & 63;
    atomicAdd(&agg[(b * SS + it * 16 + mi) * DD + e], red[x]);
  }
}

// ---------------------------------------------------------------- epilogue: LN1 -> MLP -> LN2
__global__ void __launch_bounds__(256) k_tail(const float* __restrict__ q,
                                              const float* __restrict__ agg,
                                              const float* __restrict__ mw1,
                                              const float* __restrict__ mb1,
                                              const float* __restrict__ mw2,
                                              const float* __restrict__ mb2,
                                              const float* __restrict__ g1,
                                              const float* __restrict__ b1,
                                              const float* __restrict__ g2,
                                              const float* __restrict__ b2,
                                              float* __restrict__ out) {
  __shared__ float xr[DD], hid[FF], yr[DD], stats[2];
  int bj = blockIdx.x, t = threadIdx.x;
  if (t < DD) xr[t] = q[bj * DD + t] + agg[bj * DD + t];
  __syncthreads();
  if (t == 0) {
    float mu = 0.f;
    for (int d = 0; d < DD; ++d) mu += xr[d];
    mu *= (1.0f / DD);
    float var = 0.f;
    for (int d = 0; d < DD; ++d) { float dv = xr[d] - mu; var += dv * dv; }
    var *= (1.0f / DD);
    stats[0] = mu; stats[1] = rsqrtf(var + 1e-5f);
  }
  __syncthreads();
  if (t < DD) xr[t] = (xr[t] - stats[0]) * stats[1] * g1[t] + b1[t];
  __syncthreads();
  {
    float a = mb1[t];
    for (int d = 0; d < DD; ++d) a += xr[d] * mw1[d * FF + t];
    hid[t] = gelu_erf(a);
  }
  __syncthreads();
  if (t < DD) {
    float mo = mb2[t];
    for (int k = 0; k < FF; ++k) mo += hid[k] * mw2[k * DD + t];
    yr[t] = xr[t] + mo;
  }
  __syncthreads();
  if (t == 0) {
    float mu = 0.f;
    for (int d = 0; d < DD; ++d) mu += yr[d];
    mu *= (1.0f / DD);
    float var = 0.f;
    for (int d = 0; d < DD; ++d) { float dv = yr[d] - mu; var += dv * dv; }
    var *= (1.0f / DD);
    stats[0] = mu; stats[1] = rsqrtf(var + 1e-5f);
  }
  __syncthreads();
  if (t < DD) out[bj * DD + t] = (yr[t] - stats[0]) * stats[1] * g2[t] + b2[t];
}

// ---------------------------------------------------------------- launch
extern "C" void kernel_launch(void* const* d_in, const int* in_sizes, int n_in,
                              void* d_out, int out_size, void* d_ws, size_t ws_size,
                              hipStream_t stream) {
  const float* q    = (const float*)d_in[0];
  const float* gw1  = (const float*)d_in[1];
  const float* gb1  = (const float*)d_in[2];
  const float* gw2  = (const float*)d_in[3];
  const float* gb2  = (const float*)d_in[4];
  const float* mw1  = (const float*)d_in[5];
  const float* mb1  = (const float*)d_in[6];
  const float* mw2  = (const float*)d_in[7];
  const float* mb2  = (const float*)d_in[8];
  const float* ln1g = (const float*)d_in[9];
  const float* ln1b = (const float*)d_in[10];
  const float* ln2g = (const float*)d_in[11];
  const float* ln2b = (const float*)d_in[12];
  float* out = (float*)d_out;

  // workspace layout (bytes)
  char* ws = (char*)d_ws;
  __bf16* gt   = (__bf16*)(ws);                 // 2,097,152  (GtT bf16)
  __bf16* phib = (__bf16*)(ws + 2097152);       //    32,768  (q bf16)
  float*  A1   = (float*)(ws + 2129920);        //   262,144
  float*  A2   = (float*)(ws + 2392064);        //   262,144
  __bf16* w2t  = (__bf16*)(ws + 2654208);       // 8,388,608  (W2T bf16)
  float*  agg  = (float*)(ws + 11042816);       //    65,536

  k_pack_gt <<<4096, 256, 0, stream>>>(gw2, gt);
  k_pack_phi<<<  64, 256, 0, stream>>>(q, phib);
  k_a12     <<< 256, 256, 0, stream>>>(q, gw1, A1, A2);
  k_w2      <<<2048, 256, 0, stream>>>(phib, gt, w2t);
  k_init    <<<   2,  64, 0, stream>>>(q, gb2, agg);
  k_gauge   <<< 128, 256, 0, stream>>>(A1, A2, gb1, w2t, agg);
  k_tail    <<< 256, 256, 0, stream>>>(q, agg, mw1, mb1, mw2, mb2,
                                       ln1g, ln1b, ln2g, ln2b, out);
}
